// EquivariantBlock_26542897889348
// MI455X (gfx1250) — matrix operand order repaired and checked by reading, not compiled, not run on hardware
//
#include <hip/hip_runtime.h>
#include <hip/hip_bf16.h>

// ---------------------------------------------------------------------------
// EGNN equivariant block for MI455X (gfx1250, wave32, WMMA f32_16x16x32_f16)
// ---------------------------------------------------------------------------

#define D      128
#define DMSG   273
#define KPAD   288          // DMSG padded to multiple of 32
#define WAVES  4            // waves per workgroup (128 threads)

typedef _Float16 v16h __attribute__((ext_vector_type(16)));
typedef _Float16 v4h  __attribute__((ext_vector_type(4)));
typedef float    v8f  __attribute__((ext_vector_type(8)));
typedef unsigned int v4u __attribute__((ext_vector_type(4)));

union F16Frag { v16h h; v4u q[2]; };

__device__ __forceinline__ float silu_f(float v) { return v / (1.0f + __expf(-v)); }
__device__ __forceinline__ float sigm_f(float v) { return 1.0f / (1.0f + __expf(-v)); }

// One 16x16 output tile of   D += A(16 x 32*kBlocks) * B(32*kBlocks x 16)
// A: LDS tile, row-major [16][ldA] halves.  B: f16 weights transposed [N][ldW].
// Fragment layouts per CDNA5 ISA 7.12.2:
//   A lane m=lane&15: halves = K[k0..k0+7], K[k0+16..k0+23], k0 = 32*kb + 8*hi
//   B lane: col = lane&15, halves = K[k0..k0+7], K[k0+16..k0+23], k0 = 32*kb + 8*hi
__device__ __forceinline__ v8f wmma_gemm(const _Float16* aLds, int ldA,
                                         const _Float16* wT,  int ldW,
                                         int n0, int kBlocks, int lane, v8f acc)
{
    const int m  = lane & 15;
    const int hi = lane >> 4;
    const _Float16* arow = aLds + (size_t)m * ldA + hi * 8;
    const _Float16* brow = wT + (size_t)(n0 + m) * ldW + hi * 8;
    for (int kb = 0; kb < kBlocks; ++kb) {
        F16Frag A, B;
        const _Float16* ap = arow + kb * 32;
        A.q[0] = *(const v4u*)(ap);
        A.q[1] = *(const v4u*)(ap + 16);
        const _Float16* bp = brow + kb * 32;
        B.q[0] = *(const v4u*)(bp);
        B.q[1] = *(const v4u*)(bp + 16);
        acc = __builtin_amdgcn_wmma_f32_16x16x32_f16(false, A.h, false, B.h,
                                                     (short)0, acc, false, false);
    }
    return acc;
}

// --------------------------- weight conversion -----------------------------
// W: f32 [Kin][Nout] row-major  ->  WT: f16 [Nout][Kpad], zero padded K tail.
__global__ void cvt_weight(const float* __restrict__ W, _Float16* __restrict__ WT,
                           int Kin, int Kpad, int Nout)
{
    int i = blockIdx.x * blockDim.x + threadIdx.x;
    if (i >= Kpad * Nout) return;
    int n = i / Kpad, k = i % Kpad;
    WT[i] = (k < Kin) ? (_Float16)W[(size_t)k * Nout + n] : (_Float16)0.0f;
}

// h f32 -> f16 staging (one pass over 25.6 MB; bandwidth trivial)
__global__ void cvt_h(const float* __restrict__ h, _Float16* __restrict__ hf, long n)
{
    long i = (long)blockIdx.x * blockDim.x + threadIdx.x;
    if (i < n) hf[i] = (_Float16)h[i];
}

// out[0 .. N*128)   : h_neigh accumulator -> later h_out   (zero init)
// out[N*128 .. +N*3): x_out, init with x (edge kernel atomically adds msg_x)
__global__ void init_out(float* __restrict__ out, const float* __restrict__ x, int N)
{
    long i = (long)blockIdx.x * blockDim.x + threadIdx.x;
    if (i < (long)N * D) out[i] = 0.0f;
    if (i < (long)N * 3) out[(long)N * D + i] = x[i];
}

// ------------------------------- edge kernel -------------------------------
__global__ __launch_bounds__(32 * WAVES)
void egnn_edge(const _Float16* __restrict__ hf16, const float* __restrict__ x,
               const float* __restrict__ a,
               const int* __restrict__ src, const int* __restrict__ dst,
               const _Float16* __restrict__ We1T, const float* __restrict__ b_e1,
               const _Float16* __restrict__ We2T, const float* __restrict__ b_e2,
               const float* __restrict__ W_att,  const float* __restrict__ b_att,
               const _Float16* __restrict__ Wc1T, const float* __restrict__ b_c1,
               const _Float16* __restrict__ Wc2T, const float* __restrict__ b_c2,
               const float* __restrict__ W_c3,
               float* __restrict__ out, int E, int N)
{
    __shared__ _Float16 fT[WAVES][16][KPAD];   // per-wave edge feature tile
    __shared__ _Float16 mT[WAVES][16][D];      // per-wave intermediate tile
    __shared__ float    xdir[WAVES][16][3];
    __shared__ int      sI[WAVES][16], dI[WAVES][16];

    const int lane = threadIdx.x & 31;
    const int w    = threadIdx.x >> 5;
    const int tile = blockIdx.x * WAVES + w;
    const long e0  = (long)tile * 16;
    if (e0 >= E) return;                      // wave-uniform

    // ---- build f = [h_src | h_dst | radial | a | pad] as f16 in LDS ----
    if (lane < 16) {
        long e = e0 + lane;
        int ec = (e < E) ? (int)e : 0;
        int s = src[ec], d = dst[ec];
        sI[w][lane] = s; dI[w][lane] = d;
        float dx = x[(long)s*3+0] - x[(long)d*3+0];
        float dy = x[(long)s*3+1] - x[(long)d*3+1];
        float dz = x[(long)s*3+2] - x[(long)d*3+2];
        float r  = sqrtf(dx*dx + dy*dy + dz*dz);
        float iv = 1.0f / (r + 1.0f);
        xdir[w][lane][0] = dx*iv; xdir[w][lane][1] = dy*iv; xdir[w][lane][2] = dz*iv;
        fT[w][lane][256] = (_Float16)r;
        #pragma unroll
        for (int j = 0; j < 16; ++j) fT[w][lane][257+j] = (_Float16)a[(long)ec*16 + j];
        #pragma unroll
        for (int j = DMSG; j < KPAD; ++j) fT[w][lane][j] = (_Float16)0.0f;
    }
    // h_src/h_dst rows copied as f16 in 16B chunks: 16 loads + 16 LDS stores/lane
    #pragma unroll
    for (int it = 0; it < 16; ++it) {
        int c = lane + 32 * it;               // 512 chunks of 16 bytes
        int m = c >> 5;                       // edge row 0..15
        int q = c & 31;                       // 16B chunk within 512B of h-features
        int node = (q < 16) ? sI[w][m] : dI[w][m];
        const v4u* sp = (const v4u*)(hf16 + (size_t)node * D + (size_t)(q & 15) * 8);
        *(v4u*)(&fT[w][m][q * 8]) = *sp;
    }

    const int col = lane & 15, hi = lane >> 4;

    // ================= hidden-message path ==================
    // m = silu(f @ W_e1 + b_e1)  -> mT
    for (int nt = 0; nt < 8; ++nt) {
        v8f acc = {};
        acc = wmma_gemm(&fT[w][0][0], KPAD, We1T, KPAD, nt*16, KPAD/32, lane, acc);
        float bb = b_e1[nt*16 + col];
        #pragma unroll
        for (int j = 0; j < 8; ++j)
            mT[w][j + hi*8][nt*16 + col] = (_Float16)silu_f(acc[j] + bb);
    }
    // msg_h = silu(m @ W_e2 + b_e2); att partials on the fly
    v8f   msgh[8];
    float attp[8] = {0,0,0,0,0,0,0,0};
    for (int nt = 0; nt < 8; ++nt) {
        v8f acc = {};
        acc = wmma_gemm(&mT[w][0][0], D, We2T, D, nt*16, D/32, lane, acc);
        float bb = b_e2[nt*16 + col];
        float wa = W_att[nt*16 + col];
        #pragma unroll
        for (int j = 0; j < 8; ++j) {
            float v = silu_f(acc[j] + bb);
            msgh[nt][j] = v;
            attp[j] += v * wa;
        }
    }
    // reduce att logits across the 16-lane half (rows 0-7 / 8-15)
    #pragma unroll
    for (int msk = 1; msk <= 8; msk <<= 1)
        #pragma unroll
        for (int j = 0; j < 8; ++j) attp[j] += __shfl_xor(attp[j], msk, 32);
    float ba = b_att[0];
    float att8[8];
    #pragma unroll
    for (int j = 0; j < 8; ++j) att8[j] = sigm_f(attp[j] + ba);

    // scatter h_neigh += att * msg_h
    for (int nt = 0; nt < 8; ++nt) {
        #pragma unroll
        for (int j = 0; j < 8; ++j) {
            int m = j + hi*8;
            if (e0 + m < E) {
                int d = dI[w][m];
                atomicAdd(&out[(long)d * D + nt*16 + col], att8[j] * msgh[nt][j]);
            }
        }
    }

    // ================= coordinate-message path ==================
    for (int nt = 0; nt < 8; ++nt) {
        v8f acc = {};
        acc = wmma_gemm(&fT[w][0][0], KPAD, Wc1T, KPAD, nt*16, KPAD/32, lane, acc);
        float bb = b_c1[nt*16 + col];
        #pragma unroll
        for (int j = 0; j < 8; ++j)
            mT[w][j + hi*8][nt*16 + col] = (_Float16)silu_f(acc[j] + bb);
    }
    float cp[8] = {0,0,0,0,0,0,0,0};
    for (int nt = 0; nt < 8; ++nt) {
        v8f acc = {};
        acc = wmma_gemm(&mT[w][0][0], D, Wc2T, D, nt*16, D/32, lane, acc);
        float bb = b_c2[nt*16 + col];
        float wc = W_c3[nt*16 + col];
        #pragma unroll
        for (int j = 0; j < 8; ++j) cp[j] += silu_f(acc[j] + bb) * wc;
    }
    #pragma unroll
    for (int msk = 1; msk <= 8; msk <<= 1)
        #pragma unroll
        for (int j = 0; j < 8; ++j) cp[j] += __shfl_xor(cp[j], msk, 32);

    if (col == 0) {                       // lanes 0 and 16 do the coord scatter
        float* xo = out + (long)N * D;
        #pragma unroll
        for (int j = 0; j < 8; ++j) {
            int m = j + hi*8;
            if (e0 + m < E) {
                int d = dI[w][m];
                float s = cp[j];
                atomicAdd(&xo[(long)d*3 + 0], s * xdir[w][m][0]);
                atomicAdd(&xo[(long)d*3 + 1], s * xdir[w][m][1]);
                atomicAdd(&xo[(long)d*3 + 2], s * xdir[w][m][2]);
            }
        }
    }
}

// ------------------------------- node kernel -------------------------------
// Reads h_neigh from out (h section), overwrites with h_out.
__global__ __launch_bounds__(32 * WAVES)
void egnn_node(const float* __restrict__ h, const _Float16* __restrict__ hf16,
               const _Float16* __restrict__ Wn1T, const float* __restrict__ b_n1,
               const _Float16* __restrict__ Wn2T, const float* __restrict__ b_n2,
               float* __restrict__ out, int N)
{
    __shared__ _Float16 aT[WAVES][16][256];
    __shared__ _Float16 uT[WAVES][16][D];

    const int lane = threadIdx.x & 31;
    const int w    = threadIdx.x >> 5;
    const int tile = blockIdx.x * WAVES + w;
    const long n0  = (long)tile * 16;
    if (n0 >= N) return;

    // h half: vectorized f16 row copy (8 chunks of 16B per lane)
    #pragma unroll
    for (int it = 0; it < 8; ++it) {
        int c = lane + 32 * it;               // 256 chunks
        int m = c >> 4, q = c & 15;
        long node = n0 + m; if (node >= N) node = N - 1;
        *(v4u*)(&aT[w][m][q * 8]) = *(const v4u*)(hf16 + node * D + (size_t)q * 8);
    }
    // h_neigh half: float4 load -> 4xf16 store
    #pragma unroll
    for (int it = 0; it < 16; ++it) {
        int c = lane + 32 * it;               // 512 chunks of 4 floats
        int m = c >> 5, q = c & 31;
        long node = n0 + m; if (node >= N) node = N - 1;
        float4 v = *(const float4*)(out + node * D + (size_t)q * 4);
        v4h hv; hv.x = (_Float16)v.x; hv.y = (_Float16)v.y;
        hv.z = (_Float16)v.z; hv.w = (_Float16)v.w;
        *(v4h*)(&aT[w][m][D + q * 4]) = hv;
    }

    const int col = lane & 15, hi = lane >> 4;
    for (int nt = 0; nt < 8; ++nt) {
        v8f acc = {};
        acc = wmma_gemm(&aT[w][0][0], 256, Wn1T, 256, nt*16, 256/32, lane, acc);
        float bb = b_n1[nt*16 + col];
        #pragma unroll
        for (int j = 0; j < 8; ++j)
            uT[w][j + hi*8][nt*16 + col] = (_Float16)silu_f(acc[j] + bb);
    }
    for (int nt = 0; nt < 8; ++nt) {
        v8f acc = {};
        acc = wmma_gemm(&uT[w][0][0], D, Wn2T, D, nt*16, D/32, lane, acc);
        float bb = b_n2[nt*16 + col];
        #pragma unroll
        for (int j = 0; j < 8; ++j) {
            long node = n0 + j + hi*8;
            if (node < N) {
                long o = node * D + nt*16 + col;
                out[o] = h[o] + acc[j] + bb;     // residual + bias
            }
        }
    }
}

// ------------------------------- launcher ----------------------------------
extern "C" void kernel_launch(void* const* d_in, const int* in_sizes, int n_in,
                              void* d_out, int out_size, void* d_ws, size_t ws_size,
                              hipStream_t stream)
{
    const float* h     = (const float*)d_in[0];
    const float* x     = (const float*)d_in[1];
    const float* a     = (const float*)d_in[2];
    const int*   src   = (const int*)d_in[3];
    const int*   dst   = (const int*)d_in[4];
    const float* W_e1  = (const float*)d_in[5];
    const float* b_e1  = (const float*)d_in[6];
    const float* W_e2  = (const float*)d_in[7];
    const float* b_e2  = (const float*)d_in[8];
    const float* W_att = (const float*)d_in[9];
    const float* b_att = (const float*)d_in[10];
    const float* W_n1  = (const float*)d_in[11];
    const float* b_n1  = (const float*)d_in[12];
    const float* W_n2  = (const float*)d_in[13];
    const float* b_n2  = (const float*)d_in[14];
    const float* W_c1  = (const float*)d_in[15];
    const float* b_c1  = (const float*)d_in[16];
    const float* W_c2  = (const float*)d_in[17];
    const float* b_c2  = (const float*)d_in[18];
    const float* W_c3  = (const float*)d_in[19];

    const int N = in_sizes[0] / D;
    const int E = in_sizes[3];

    // workspace layout: f16 transposed weights (~304 KB) + f16 copy of h
    char* ws = (char*)d_ws;
    _Float16* We1T = (_Float16*)(ws + 0);        // 128*288*2 = 73728
    _Float16* Wc1T = (_Float16*)(ws + 73728);
    _Float16* We2T = (_Float16*)(ws + 147456);   // 128*128*2 = 32768
    _Float16* Wc2T = (_Float16*)(ws + 180224);
    _Float16* Wn1T = (_Float16*)(ws + 212992);   // 128*256*2 = 65536
    _Float16* Wn2T = (_Float16*)(ws + 278528);
    _Float16* hf16 = (_Float16*)(ws + 311296);   // N*128*2 bytes

    auto cv = [&](const float* W, _Float16* WT, int Kin, int Kpad, int Nout) {
        int tot = Kpad * Nout;
        cvt_weight<<<(tot + 255) / 256, 256, 0, stream>>>(W, WT, Kin, Kpad, Nout);
    };
    cv(W_e1, We1T, DMSG, KPAD, D);
    cv(W_c1, Wc1T, DMSG, KPAD, D);
    cv(W_e2, We2T, D, D, D);
    cv(W_c2, Wc2T, D, D, D);
    cv(W_n1, Wn1T, 2 * D, 2 * D, D);
    cv(W_n2, Wn2T, D, D, D);

    long tot = (long)N * D;
    cvt_h<<<(int)((tot + 255) / 256), 256, 0, stream>>>(h, hf16, tot);

    float* out = (float*)d_out;
    init_out<<<(int)((tot + 255) / 256), 256, 0, stream>>>(out, x, N);

    int etiles = (E + 15) / 16;
    egnn_edge<<<(etiles + WAVES - 1) / WAVES, 32 * WAVES, 0, stream>>>(
        hf16, x, a, src, dst, We1T, b_e1, We2T, b_e2, W_att, b_att,
        Wc1T, b_c1, Wc2T, b_c2, W_c3, out, E, N);

    int ntiles = (N + 15) / 16;
    egnn_node<<<(ntiles + WAVES - 1) / WAVES, 32 * WAVES, 0, stream>>>(
        h, hf16, Wn1T, b_n1, Wn2T, b_n2, out, N);
}
